// FusedMoe_89644557402996
// MI455X (gfx1250) — compile-verified, matrix operand order checked by reading, not run on hardware
//
#include <hip/hip_runtime.h>
#include <stdint.h>

// ---------------- problem constants (match reference) ----------------
#define TT    8192   // tokens
#define KK    2048   // hidden
#define NN    2048   // intermediate
#define EE    16     // experts
#define TOPKK 2      // top-k
// ---------------- tiling ----------------
#define MT        32    // token rows per block (two WMMA M subtiles)
#define NT_BLOCK  128   // output cols per block (4 waves * 32)
#define KC        512   // K-chunk staged in LDS (32KB bf16 at MT=32)
#define THREADS   128   // 4 waves of 32

typedef __bf16 v16bf __attribute__((ext_vector_type(16)));
typedef float  v8f   __attribute__((ext_vector_type(8)));

// fragment viewed as raw 16-byte halves: lets us fill it with two b128 loads
// and hand it to WMMA without per-element shuffles.
union Frag16 {
  v16bf v;
  uint4 q[2];
};

__device__ __forceinline__ uint32_t bfround(uint32_t u) {
  return (u + 0x7FFFu + ((u >> 16) & 1u)) >> 16;
}
__device__ __forceinline__ uint32_t pack_bf16x2(float lo, float hi) {
  union { float f; uint32_t u; } a, b;
  a.f = lo; b.f = hi;
  return bfround(a.u) | (bfround(b.u) << 16);
}
__device__ __forceinline__ __bf16 f2bf(float x) {
  union { float f; uint32_t u; } v; v.f = x;
  union { uint16_t s; __bf16 b; } o; o.s = (uint16_t)bfround(v.u);
  return o.b;
}

// ---------------- utility kernels ----------------
__global__ void zero_i32_kernel(int* p, int n) {
  int i = blockIdx.x * blockDim.x + threadIdx.x;
  if (i < n) p[i] = 0;
}
__global__ void zero_f32_kernel(float* p, long long n) {
  long long i = (long long)blockIdx.x * blockDim.x + threadIdx.x;
  if (i < n) p[i] = 0.0f;
}
// streaming fp32 -> bf16 conversion, 4 elements per thread (one-shot pass)
__global__ void cvt_bf16_kernel(const float* __restrict__ in,
                                uint32_t* __restrict__ out, long long n4) {
  long long i = (long long)blockIdx.x * blockDim.x + threadIdx.x;
  if (i >= n4) return;
  float4 v = reinterpret_cast<const float4*>(in)[i];
  uint2 p;
  p.x = pack_bf16x2(v.x, v.y);
  p.y = pack_bf16x2(v.z, v.w);
  reinterpret_cast<uint2*>(out)[i] = p;
}

// ---------------- router: softmax + top-2 + expert counting ----------------
__global__ void router_kernel(const float* __restrict__ score,
                              const int* __restrict__ e_map,
                              int* __restrict__ topk_ids,
                              float* __restrict__ topk_w,
                              int* __restrict__ counts) {
  int t = blockIdx.x * blockDim.x + threadIdx.x;
  if (t >= TT) return;
  float s[EE];
  float m = -3.4e38f;
  #pragma unroll
  for (int e = 0; e < EE; ++e) { s[e] = score[t * EE + e]; m = fmaxf(m, s[e]); }
  float sum = 0.f;
  #pragma unroll
  for (int e = 0; e < EE; ++e) { s[e] = __expf(s[e] - m); sum += s[e]; }
  float inv = 1.f / sum;
  unsigned used = 0u;
  #pragma unroll
  for (int k = 0; k < TOPKK; ++k) {
    int best = 0; float bv = -1.f;
    #pragma unroll
    for (int e = 0; e < EE; ++e)
      if (!((used >> e) & 1u) && s[e] > bv) { bv = s[e]; best = e; }
    used |= (1u << best);
    int ex = e_map[best];
    topk_ids[t * TOPKK + k] = ex;
    topk_w[t * TOPKK + k]   = bv * inv;
    atomicAdd(&counts[ex], 1);
  }
}

__global__ void prefix_kernel(const int* __restrict__ counts,
                              int* __restrict__ offsets) {
  if (threadIdx.x == 0) {
    int acc = 0;
    for (int e = 0; e < EE; ++e) { offsets[e] = acc; acc += counts[e]; }
  }
}

__global__ void scatter_kernel(const int* __restrict__ topk_ids,
                               const float* __restrict__ topk_w,
                               const int* __restrict__ offsets,
                               int* __restrict__ cursor,
                               int* __restrict__ disp_tok,
                               float* __restrict__ disp_gate) {
  int t = blockIdx.x * blockDim.x + threadIdx.x;
  if (t >= TT) return;
  #pragma unroll
  for (int k = 0; k < TOPKK; ++k) {
    int ex = topk_ids[t * TOPKK + k];
    int pos = atomicAdd(&cursor[ex], 1);
    int slot = offsets[ex] + pos;
    disp_tok[slot]  = t;
    disp_gate[slot] = topk_w[t * TOPKK + k];
  }
}

// B fragment: two raw b128 loads from a pre-converted bf16 weight row.
// elements 0..7 = k[koff..koff+7], 8..15 = k[16+koff..16+koff+7]  (ISA layout)
__device__ __forceinline__ v16bf load_b_frag_bf16(const __bf16* __restrict__ row,
                                                  int kbase, int koff) {
  Frag16 f;
  f.q[0] = *reinterpret_cast<const uint4*>(row + kbase + koff);
  f.q[1] = *reinterpret_cast<const uint4*>(row + kbase + 16 + koff);
  return f.v;
}

// ---------------- GEMM1: act = silu(a@w1_gate^T) * (a@w1_up^T) ----------------
__global__ __launch_bounds__(THREADS)
void gemm1_kernel(const __bf16* __restrict__ abf,
                  const __bf16* __restrict__ w1bf,
                  const int* __restrict__ disp_tok,
                  const int* __restrict__ offsets,
                  const int* __restrict__ counts,
                  __bf16* __restrict__ act) {
  const int e  = blockIdx.z;
  const int mt = blockIdx.y;
  const int count = counts[e];
  if (mt * MT >= count) return;
  const int base_slot  = offsets[e] + mt * MT;
  const int rows_valid = min(MT, count - mt * MT);

  __shared__ __bf16 Alds[MT][KC];
  __shared__ int    s_tok[MT];

  const int tid = threadIdx.x;
  if (tid < MT) s_tok[tid] = (tid < rows_valid) ? disp_tok[base_slot + tid] : -1;

  const int lane = tid & 31;
  const int wave = tid >> 5;
  const int l16  = lane & 15;
  const int koff = (lane >> 4) * 8;
  const int ng   = blockIdx.x * NT_BLOCK + wave * 32;  // gate-column base

  const __bf16* __restrict__ w1e = w1bf + (size_t)e * (2 * NN) * KK;
  const __bf16* __restrict__ wg0 = w1e + (size_t)(ng      + l16) * KK;
  const __bf16* __restrict__ wg1 = w1e + (size_t)(ng + 16 + l16) * KK;
  const __bf16* __restrict__ wu0 = w1e + (size_t)(ng + NN      + l16) * KK;
  const __bf16* __restrict__ wu1 = w1e + (size_t)(ng + NN + 16 + l16) * KK;

  // accumulators: [m-subtile][n-subtile] for gate and up
  v8f ag00 = {}, ag01 = {}, ag10 = {}, ag11 = {};
  v8f au00 = {}, au01 = {}, au10 = {}, au11 = {};

  for (int kc = 0; kc < KK; kc += KC) {
    __syncthreads();
    // stage 32 gathered bf16 token rows into LDS (pure 16B copies)
    for (int idx = tid; idx < MT * (KC / 8); idx += THREADS) {
      int r  = idx / (KC / 8);
      int c8 = (idx % (KC / 8)) * 8;
      int tok = s_tok[r];
      uint4 v = make_uint4(0u, 0u, 0u, 0u);
      if (tok >= 0)
        v = *reinterpret_cast<const uint4*>(&abf[(size_t)tok * KK + kc + c8]);
      *reinterpret_cast<uint4*>(&Alds[r][c8]) = v;
    }
    __syncthreads();

    for (int k0 = 0; k0 < KC; k0 += 32) {
      const int kg = kc + k0;
      // A fragments for the two M subtiles, straight b128 LDS loads
      Frag16 a0, a1;
      a0.q[0] = *reinterpret_cast<const uint4*>(&Alds[l16][k0 + koff]);
      a0.q[1] = *reinterpret_cast<const uint4*>(&Alds[l16][k0 + 16 + koff]);
      a1.q[0] = *reinterpret_cast<const uint4*>(&Alds[16 + l16][k0 + koff]);
      a1.q[1] = *reinterpret_cast<const uint4*>(&Alds[16 + l16][k0 + 16 + koff]);

      if (kg + 32 < KK) {
        __builtin_prefetch(wg0 + kg + 32, 0, 1);
        __builtin_prefetch(wu0 + kg + 32, 0, 1);
      }
      v16bf bg0 = load_b_frag_bf16(wg0, kg, koff);
      v16bf bg1 = load_b_frag_bf16(wg1, kg, koff);
      v16bf bu0 = load_b_frag_bf16(wu0, kg, koff);
      v16bf bu1 = load_b_frag_bf16(wu1, kg, koff);

      ag00 = __builtin_amdgcn_wmma_f32_16x16x32_bf16(false, a0.v, false, bg0, (short)0, ag00, false, false);
      ag01 = __builtin_amdgcn_wmma_f32_16x16x32_bf16(false, a0.v, false, bg1, (short)0, ag01, false, false);
      ag10 = __builtin_amdgcn_wmma_f32_16x16x32_bf16(false, a1.v, false, bg0, (short)0, ag10, false, false);
      ag11 = __builtin_amdgcn_wmma_f32_16x16x32_bf16(false, a1.v, false, bg1, (short)0, ag11, false, false);
      au00 = __builtin_amdgcn_wmma_f32_16x16x32_bf16(false, a0.v, false, bu0, (short)0, au00, false, false);
      au01 = __builtin_amdgcn_wmma_f32_16x16x32_bf16(false, a0.v, false, bu1, (short)0, au01, false, false);
      au10 = __builtin_amdgcn_wmma_f32_16x16x32_bf16(false, a1.v, false, bu0, (short)0, au10, false, false);
      au11 = __builtin_amdgcn_wmma_f32_16x16x32_bf16(false, a1.v, false, bu1, (short)0, au11, false, false);
    }
  }

  // fused silu(gate) * up, store bf16 activations
  #pragma unroll
  for (int ms = 0; ms < 2; ++ms) {
    #pragma unroll
    for (int nt = 0; nt < 2; ++nt) {
      v8f g = ms ? (nt ? ag11 : ag10) : (nt ? ag01 : ag00);
      v8f u = ms ? (nt ? au11 : au10) : (nt ? au01 : au00);
      int col = ng + nt * 16 + l16;
      #pragma unroll
      for (int r = 0; r < 8; ++r) {
        int m = ms * 16 + r + (lane >> 4) * 8;  // C layout: VGPR r -> M=r / 8+r
        if (m < rows_valid) {
          float gv = g[r];
          float sv = gv * __builtin_amdgcn_rcpf(1.f + __expf(-gv));
          act[(size_t)(base_slot + m) * NN + col] = f2bf(sv * u[r]);
        }
      }
    }
  }
}

// ---------------- GEMM2: out += gate * (act @ w2^T) ----------------
__global__ __launch_bounds__(THREADS)
void gemm2_kernel(const __bf16* __restrict__ act,
                  const __bf16* __restrict__ w2bf,
                  const int* __restrict__ disp_tok,
                  const float* __restrict__ disp_gate,
                  const int* __restrict__ offsets,
                  const int* __restrict__ counts,
                  float* __restrict__ out) {
  const int e  = blockIdx.z;
  const int mt = blockIdx.y;
  const int count = counts[e];
  if (mt * MT >= count) return;
  const int base_slot  = offsets[e] + mt * MT;
  const int rows_valid = min(MT, count - mt * MT);

  __shared__ __bf16 Hlds[MT][KC];
  __shared__ int    s_tok[MT];
  __shared__ float  s_gate[MT];

  const int tid = threadIdx.x;
  if (tid < MT) {
    s_tok[tid]  = (tid < rows_valid) ? disp_tok[base_slot + tid] : 0;
    s_gate[tid] = (tid < rows_valid) ? disp_gate[base_slot + tid] : 0.f;
  }

  const int lane = tid & 31;
  const int wave = tid >> 5;
  const int l16  = lane & 15;
  const int koff = (lane >> 4) * 8;
  const int cb   = blockIdx.x * NT_BLOCK + wave * 32;  // output-column base

  const __bf16* __restrict__ w2e = w2bf + (size_t)e * KK * NN;
  const __bf16* __restrict__ wr0 = w2e + (size_t)(cb      + l16) * NN;
  const __bf16* __restrict__ wr1 = w2e + (size_t)(cb + 16 + l16) * NN;

  v8f a00 = {}, a01 = {}, a10 = {}, a11 = {};

  for (int nc = 0; nc < NN; nc += KC) {
    __syncthreads();
    for (int idx = tid; idx < MT * (KC / 8); idx += THREADS) {
      int r  = idx / (KC / 8);
      int c8 = (idx % (KC / 8)) * 8;
      uint4 v = make_uint4(0u, 0u, 0u, 0u);
      if (r < rows_valid)
        v = *reinterpret_cast<const uint4*>(&act[(size_t)(base_slot + r) * NN + nc + c8]);
      *reinterpret_cast<uint4*>(&Hlds[r][c8]) = v;
    }
    __syncthreads();

    for (int k0 = 0; k0 < KC; k0 += 32) {
      const int ngl = nc + k0;
      Frag16 h0, h1;
      h0.q[0] = *reinterpret_cast<const uint4*>(&Hlds[l16][k0 + koff]);
      h0.q[1] = *reinterpret_cast<const uint4*>(&Hlds[l16][k0 + 16 + koff]);
      h1.q[0] = *reinterpret_cast<const uint4*>(&Hlds[16 + l16][k0 + koff]);
      h1.q[1] = *reinterpret_cast<const uint4*>(&Hlds[16 + l16][k0 + 16 + koff]);

      if (ngl + 32 < NN) __builtin_prefetch(wr0 + ngl + 32, 0, 1);
      v16bf b0 = load_b_frag_bf16(wr0, ngl, koff);
      v16bf b1 = load_b_frag_bf16(wr1, ngl, koff);

      a00 = __builtin_amdgcn_wmma_f32_16x16x32_bf16(false, h0.v, false, b0, (short)0, a00, false, false);
      a01 = __builtin_amdgcn_wmma_f32_16x16x32_bf16(false, h0.v, false, b1, (short)0, a01, false, false);
      a10 = __builtin_amdgcn_wmma_f32_16x16x32_bf16(false, h1.v, false, b0, (short)0, a10, false, false);
      a11 = __builtin_amdgcn_wmma_f32_16x16x32_bf16(false, h1.v, false, b1, (short)0, a11, false, false);
    }
  }

  #pragma unroll
  for (int ms = 0; ms < 2; ++ms) {
    #pragma unroll
    for (int nt = 0; nt < 2; ++nt) {
      v8f acc = ms ? (nt ? a11 : a10) : (nt ? a01 : a00);
      int col = cb + nt * 16 + l16;
      #pragma unroll
      for (int r = 0; r < 8; ++r) {
        int m = ms * 16 + r + (lane >> 4) * 8;
        if (m < rows_valid) {
          atomicAdd(&out[(size_t)s_tok[m] * KK + col], acc[r] * s_gate[m]);
        }
      }
    }
  }
}

// ---------------- host-side launch ----------------
extern "C" void kernel_launch(void* const* d_in, const int* in_sizes, int n_in,
                              void* d_out, int out_size, void* d_ws, size_t ws_size,
                              hipStream_t stream) {
  (void)in_sizes; (void)n_in; (void)ws_size;
  const float* a     = (const float*)d_in[0];
  const float* w1    = (const float*)d_in[1];
  const float* w2    = (const float*)d_in[2];
  const float* score = (const float*)d_in[3];
  const int*   e_map = (const int*)d_in[6];
  float*       out   = (float*)d_out;

  // workspace carve-up (256B aligned)
  char* ws = (char*)d_ws;
  size_t off = 0;
  auto carve = [&](size_t bytes) -> void* {
    off = (off + 255) & ~(size_t)255;
    void* p = ws + off;
    off += bytes;
    return p;
  };
  int*    counts    = (int*)carve(EE * sizeof(int) * 2);     // counts + cursor
  int*    cursor    = counts + EE;
  int*    offsets   = (int*)carve(EE * sizeof(int));
  int*    topk_ids  = (int*)carve((size_t)TT * TOPKK * sizeof(int));
  float*  topk_w    = (float*)carve((size_t)TT * TOPKK * sizeof(float));
  int*    disp_tok  = (int*)carve((size_t)TT * TOPKK * sizeof(int));
  float*  disp_gate = (float*)carve((size_t)TT * TOPKK * sizeof(float));
  __bf16* actbuf    = (__bf16*)carve((size_t)TT * TOPKK * NN * sizeof(__bf16));
  __bf16* abf       = (__bf16*)carve((size_t)TT * KK * sizeof(__bf16));
  __bf16* w1bf      = (__bf16*)carve((size_t)EE * 2 * NN * KK * sizeof(__bf16));
  __bf16* w2bf      = (__bf16*)carve((size_t)EE * KK * NN * sizeof(__bf16));

  // 1) zero counters and output accumulator
  zero_i32_kernel<<<1, 64, 0, stream>>>(counts, 2 * EE);
  {
    long long n = (long long)out_size;
    int blocks = (int)((n + 255) / 256);
    zero_f32_kernel<<<blocks, 256, 0, stream>>>(out, n);
  }
  // 2) one-shot fp32 -> bf16 pre-conversion (a, w1, w2)
  {
    long long n4;
    n4 = (long long)TT * KK / 4;
    cvt_bf16_kernel<<<(int)((n4 + 255) / 256), 256, 0, stream>>>(a, (uint32_t*)abf, n4);
    n4 = (long long)EE * 2 * NN * KK / 4;
    cvt_bf16_kernel<<<(int)((n4 + 255) / 256), 256, 0, stream>>>(w1, (uint32_t*)w1bf, n4);
    n4 = (long long)EE * KK * NN / 4;
    cvt_bf16_kernel<<<(int)((n4 + 255) / 256), 256, 0, stream>>>(w2, (uint32_t*)w2bf, n4);
  }
  // 3) router: softmax + top-2 + expert counts
  router_kernel<<<TT / 256, 256, 0, stream>>>(score, e_map, topk_ids, topk_w, counts);
  // 4) exclusive prefix sum over expert counts
  prefix_kernel<<<1, 32, 0, stream>>>(counts, offsets);
  // 5) scatter tokens into contiguous per-expert slot lists
  scatter_kernel<<<TT / 256, 256, 0, stream>>>(topk_ids, topk_w, offsets, cursor,
                                               disp_tok, disp_gate);
  // 6) GEMM1 + fused SiLU*mul -> bf16 activations
  {
    dim3 grid(NN / NT_BLOCK, TT / MT, EE);
    gemm1_kernel<<<grid, THREADS, 0, stream>>>(abf, w1bf, disp_tok, offsets, counts, actbuf);
  }
  // 7) GEMM2, gate-scaled atomic accumulation into out
  {
    dim3 grid(KK / NT_BLOCK, TT / MT, EE);
    gemm2_kernel<<<grid, THREADS, 0, stream>>>(actbuf, w2bf, disp_tok, disp_gate,
                                               offsets, counts, out);
  }
}
